// ModelQuantLinear_6382321401940
// MI455X (gfx1250) — 8-bit matrix operand order repaired and checked by reading, compiled, not run on hardware
//
#include <hip/hip_runtime.h>
#include <stdint.h>

typedef __attribute__((ext_vector_type(8))) int v8i;

__global__ void init_scalars(unsigned* s) {
    if (threadIdx.x < 3) s[threadIdx.x] = 0u;
}

__global__ void absmax_kernel(const float4* __restrict__ x, size_t n4,
                              unsigned* __restrict__ out_bits) {
    __shared__ float red[256];
    float m = 0.0f;
    for (size_t i = (size_t)blockIdx.x * blockDim.x + threadIdx.x;
         i < n4; i += (size_t)gridDim.x * blockDim.x) {
        float4 v = x[i];
        m = fmaxf(m, fmaxf(fmaxf(fabsf(v.x), fabsf(v.y)),
                           fmaxf(fabsf(v.z), fabsf(v.w))));
    }
    red[threadIdx.x] = m;
    __syncthreads();
    #pragma unroll
    for (int s = 128; s > 0; s >>= 1) {
        if ((int)threadIdx.x < s)
            red[threadIdx.x] = fmaxf(red[threadIdx.x], red[threadIdx.x + s]);
        __syncthreads();
    }
    if (threadIdx.x == 0) atomicMax(out_bits, __float_as_uint(red[0]));
}

__global__ void quant_kernel(const float4* __restrict__ x, char4* __restrict__ q,
                             int n4, const unsigned* __restrict__ max_bits,
                             float qmax, float qmin) {
    int i = blockIdx.x * blockDim.x + threadIdx.x;
    if (i >= n4) return;
    float s   = fmaxf(__uint_as_float(*max_bits) / qmax, 1e-8f);
    float inv = 1.0f / s;
    float4 v = x[i];
    char4 o;
    o.x = (signed char)(int)fminf(fmaxf(rintf(v.x * inv), qmin), qmax);
    o.y = (signed char)(int)fminf(fmaxf(rintf(v.y * inv), qmin), qmax);
    o.z = (signed char)(int)fminf(fmaxf(rintf(v.z * inv), qmin), qmax);
    o.w = (signed char)(int)fminf(fmaxf(rintf(v.w * inv), qmin), qmax);
    q[i] = o;
}

#define BLK_M   128
#define BLK_N   128
#define KSTEP   64
#define LDS_STR 80

__global__ void __launch_bounds__(256)
gemm_i8_wmma(const int8_t* __restrict__ Aq, const int8_t* __restrict__ Bq,
             const float* __restrict__ bias, float* __restrict__ Y,
             const unsigned* __restrict__ in_bits, unsigned* __restrict__ y_bits,
             int M, int N, int K) {
    __shared__ uint8_t smA[2][BLK_M * LDS_STR];
    __shared__ uint8_t smB[2][BLK_N * LDS_STR];
    __shared__ float   red[256];

    const int tid  = threadIdx.x;
    const int lane = tid & 31;
    const int wave = tid >> 5;
    const int half = lane >> 4;
    const int l16  = lane & 15;
    const int waveM = wave >> 1;
    const int waveN = wave & 1;

    const int blockM = blockIdx.x * BLK_M;
    const int blockN = blockIdx.y * BLK_N;

    const int ca0 = tid,        ca1 = tid + 256;
    const int ar0 = ca0 >> 2,   ak0 = (ca0 & 3) * 16;
    const int ar1 = ca1 >> 2,   ak1 = (ca1 & 3) * 16;

    const uint32_t ldsA[2] = { (uint32_t)(uintptr_t)&smA[0][0],
                               (uint32_t)(uintptr_t)&smA[1][0] };
    const uint32_t ldsB[2] = { (uint32_t)(uintptr_t)&smB[0][0],
                               (uint32_t)(uintptr_t)&smB[1][0] };

    const int8_t* Ab = Aq + (size_t)blockM * K;
    const int8_t* Bb = Bq + (size_t)blockN * K;

    auto issue_tile = [&](int buf, int k0) {
        uint64_t ga0 = (uint64_t)(uintptr_t)(Ab + (size_t)ar0 * K + k0 + ak0);
        uint64_t ga1 = (uint64_t)(uintptr_t)(Ab + (size_t)ar1 * K + k0 + ak1);
        uint64_t gb0 = (uint64_t)(uintptr_t)(Bb + (size_t)ar0 * K + k0 + ak0);
        uint64_t gb1 = (uint64_t)(uintptr_t)(Bb + (size_t)ar1 * K + k0 + ak1);
        uint32_t la0 = ldsA[buf] + ar0 * LDS_STR + ak0;
        uint32_t la1 = ldsA[buf] + ar1 * LDS_STR + ak1;
        uint32_t lb0 = ldsB[buf] + ar0 * LDS_STR + ak0;
        uint32_t lb1 = ldsB[buf] + ar1 * LDS_STR + ak1;
        asm volatile("global_load_async_to_lds_b128 %0, %1, off"
                     :: "v"(la0), "v"(ga0) : "memory");
        asm volatile("global_load_async_to_lds_b128 %0, %1, off"
                     :: "v"(la1), "v"(ga1) : "memory");
        asm volatile("global_load_async_to_lds_b128 %0, %1, off"
                     :: "v"(lb0), "v"(gb0) : "memory");
        asm volatile("global_load_async_to_lds_b128 %0, %1, off"
                     :: "v"(lb1), "v"(gb1) : "memory");
    };

    v8i acc[2][4] = {};

    issue_tile(0, 0);
    asm volatile("s_wait_asynccnt 0x0" ::: "memory");
    __syncthreads();

    for (int k0 = 0; k0 < K; k0 += KSTEP) {
        const int cur = (k0 >> 6) & 1;
        if (k0 + KSTEP < K) issue_tile(cur ^ 1, k0 + KSTEP);

        const uint8_t* Abase = &smA[cur][0];
        const uint8_t* Bbase = &smB[cur][0];

        v8i a[2], b[4];
        #pragma unroll
        for (int mt = 0; mt < 2; ++mt) {
            const uint8_t* r = Abase + (waveM * 32 + mt * 16 + l16) * LDS_STR;
            #pragma unroll
            for (int c = 0; c < 4; ++c) {
                uint2 d = *(const uint2*)(r + 32 * (c >> 1) + 16 * half + 8 * (c & 1));
                a[mt][2 * c]     = (int)d.x;
                a[mt][2 * c + 1] = (int)d.y;
            }
        }
        #pragma unroll
        for (int nt = 0; nt < 4; ++nt) {
            const uint8_t* r = Bbase + (waveN * 64 + nt * 16 + l16) * LDS_STR;
            uint4 b0 = *(const uint4*)(r + 16 * half);
            uint4 b1 = *(const uint4*)(r + 32 + 16 * half);
            b[nt][0] = (int)b0.x; b[nt][1] = (int)b0.y;
            b[nt][2] = (int)b0.z; b[nt][3] = (int)b0.w;
            b[nt][4] = (int)b1.x; b[nt][5] = (int)b1.y;
            b[nt][6] = (int)b1.z; b[nt][7] = (int)b1.w;
        }
        #pragma unroll
        for (int mt = 0; mt < 2; ++mt)
            #pragma unroll
            for (int nt = 0; nt < 4; ++nt)
                acc[mt][nt] = __builtin_amdgcn_wmma_i32_16x16x64_iu8(
                    true, a[mt], true, b[nt], acc[mt][nt], false, false);

        asm volatile("s_wait_asynccnt 0x0" ::: "memory");
        __syncthreads();
    }

    const float s_in = fmaxf(__uint_as_float(in_bits[0]) / 127.0f, 1e-8f);
    const float s_w  = fmaxf(__uint_as_float(in_bits[1]) /   7.0f, 1e-8f);
    const float s    = s_in * s_w;

    const int m_base = blockM + waveM * 32;
    const int n_base = blockN + waveN * 64;

    float bdq[4];
    #pragma unroll
    for (int nt = 0; nt < 4; ++nt)
        bdq[nt] = rintf(bias[n_base + nt * 16 + l16] / s) * s;

    float lmax = 0.0f;
    #pragma unroll
    for (int mt = 0; mt < 2; ++mt) {
        #pragma unroll
        for (int nt = 0; nt < 4; ++nt) {
            const int col = n_base + nt * 16 + l16;
            #pragma unroll
            for (int j = 0; j < 8; ++j) {
                const int row = m_base + mt * 16 + j + 8 * half;
                float y = (float)acc[mt][nt][j] * s + bdq[nt];
                Y[(size_t)row * N + col] = y;
                lmax = fmaxf(lmax, fabsf(y));
            }
        }
    }

    red[threadIdx.x] = lmax;
    __syncthreads();
    #pragma unroll
    for (int st = 128; st > 0; st >>= 1) {
        if ((int)threadIdx.x < st)
            red[threadIdx.x] = fmaxf(red[threadIdx.x], red[threadIdx.x + st]);
        __syncthreads();
    }
    if (threadIdx.x == 0) atomicMax(y_bits, __float_as_uint(red[0]));
}

__global__ void quant_out_kernel(float4* __restrict__ y, int n4,
                                 const unsigned* __restrict__ y_bits) {
    int i = blockIdx.x * blockDim.x + threadIdx.x;
    if (i >= n4) return;
    float s   = fmaxf(__uint_as_float(*y_bits) / 127.0f, 1e-8f);
    float inv = 1.0f / s;
    float4 v = y[i];
    v.x = fminf(fmaxf(rintf(v.x * inv), -128.0f), 127.0f) * s;
    v.y = fminf(fmaxf(rintf(v.y * inv), -128.0f), 127.0f) * s;
    v.z = fminf(fmaxf(rintf(v.z * inv), -128.0f), 127.0f) * s;
    v.w = fminf(fmaxf(rintf(v.w * inv), -128.0f), 127.0f) * s;
    y[i] = v;
}

extern "C" void kernel_launch(void* const* d_in, const int* in_sizes, int n_in,
                              void* d_out, int out_size, void* d_ws, size_t ws_size,
                              hipStream_t stream) {
    const float* inp = (const float*)d_in[0];
    const float* W   = (const float*)d_in[1];
    const float* b   = (const float*)d_in[2];
    float* out = (float*)d_out;

    const int K = 4096;
    const int N = in_sizes[2];
    const int M = in_sizes[0] / K;

    unsigned char* ws = (unsigned char*)d_ws;
    unsigned* scal = (unsigned*)ws;
    int8_t* Xq = (int8_t*)(ws + 256);
    int8_t* Wq = (int8_t*)(ws + 256 + (size_t)M * K);

    const size_t nx4 = (size_t)M * K / 4;
    const size_t nw4 = (size_t)N * K / 4;
    const size_t ny4 = (size_t)M * N / 4;

    init_scalars<<<1, 64, 0, stream>>>(scal);
    absmax_kernel<<<2048, 256, 0, stream>>>((const float4*)inp, nx4, &scal[0]);
    absmax_kernel<<<1024, 256, 0, stream>>>((const float4*)W,   nw4, &scal[1]);
    quant_kernel<<<(int)((nx4 + 255) / 256), 256, 0, stream>>>(
        (const float4*)inp, (char4*)Xq, (int)nx4, &scal[0], 127.0f, -128.0f);
    quant_kernel<<<(int)((nw4 + 255) / 256), 256, 0, stream>>>(
        (const float4*)W, (char4*)Wq, (int)nw4, &scal[1], 7.0f, -8.0f);

    dim3 grid(M / BLK_M, N / BLK_N);
    gemm_i8_wmma<<<grid, 256, 0, stream>>>(Xq, Wq, b, out,
                                           scal, &scal[2], M, N, K);

    quant_out_kernel<<<(int)((ny4 + 255) / 256), 256, 0, stream>>>(
        (float4*)out, (int)ny4, &scal[2]);
}
